// Decoder_34479997452599
// MI455X (gfx1250) — matrix operand order repaired and checked by reading, not compiled, not run on hardware
//
#include <hip/hip_runtime.h>
#include <hip/hip_bf16.h>
#include <math.h>

// ---------------------------------------------------------------------------
// Problem constants (from reference): V=32000, E=512, H=1024, L=2, B=64, T=64
// Steps = T-1 = 63, rows R = 63*64 = 4032, gate width G = 4H = 4096.
// ---------------------------------------------------------------------------
#define VSZ   32000
#define ESZ   512
#define HSZ   1024
#define GSZ   4096
#define BSZ   64
#define TSTEPS 63
#define ROWS  (TSTEPS * BSZ)     // 4032
#define NCB   63                 // projection column blocks of 512
#define PROJ_COLS 512

typedef __bf16 bf16_t;
typedef __attribute__((ext_vector_type(16))) __bf16 v16bf;
typedef __attribute__((ext_vector_type(8)))  float  v8f;
typedef unsigned int u32;
typedef __attribute__((ext_vector_type(4))) u32 u32x4;
typedef __attribute__((ext_vector_type(4))) int v4i;

// ---------------------------------------------------------------------------
// CDNA5 async global->LDS staging (ASYNCcnt path), with sync b128 fallback.
// Probe result (round 2): builtin exists with signature
//   void __builtin_amdgcn_global_load_async_to_lds_b128(
//        v4i addrspace(1)* src, v4i addrspace(3)* dst, imm int off, imm int cpol)
// ---------------------------------------------------------------------------
#if defined(__has_builtin)
#if __has_builtin(__builtin_amdgcn_global_load_async_to_lds_b128)
#define HAVE_ASYNC_LDS 1
#endif
#endif
#ifndef HAVE_ASYNC_LDS
#define HAVE_ASYNC_LDS 0
#endif

#if HAVE_ASYNC_LDS
typedef __attribute__((address_space(1))) v4i as1_v4i;
typedef __attribute__((address_space(3))) v4i as3_v4i;
__device__ inline void lds_copy_b128(const bf16_t* g, bf16_t* l) {
  __builtin_amdgcn_global_load_async_to_lds_b128((as1_v4i*)g, (as3_v4i*)l, 0, 0);
}
#if __has_builtin(__builtin_amdgcn_s_wait_asynccnt)
#define ASYNC_WAIT() __builtin_amdgcn_s_wait_asynccnt(0)
#else
#define ASYNC_WAIT() asm volatile("s_wait_asynccnt 0x0" ::: "memory")
#endif
#else
__device__ inline void lds_copy_b128(const bf16_t* g, bf16_t* l) {
  *(u32x4*)l = *(const u32x4*)g;
}
#define ASYNC_WAIT() ((void)0)
#endif

// ---------------------------------------------------------------------------
// Elementwise fp32 -> bf16 cast (weight prep: halves bytes -> whole model is
// L2-resident; enables v_wmma_f32_16x16x32_bf16)
// ---------------------------------------------------------------------------
__global__ __launch_bounds__(256) void k_cast_bf16(const float* __restrict__ src,
                                                   bf16_t* __restrict__ dst, int n) {
  int i = blockIdx.x * 256 + threadIdx.x;
  if (i < n) dst[i] = (bf16_t)src[i];
}

// ---------------------------------------------------------------------------
// Embedding gather with pad-zero mask, fp32 -> bf16
// ---------------------------------------------------------------------------
__global__ __launch_bounds__(256) void k_embed(const int* __restrict__ tokens,
                                               const float* __restrict__ emb,
                                               bf16_t* __restrict__ x, int n) {
  int i = blockIdx.x * 256 + threadIdx.x;
  if (i >= n) return;
  int tb = i >> 9;          // row in (63*64)
  int e  = i & (ESZ - 1);
  int tok = tokens[tb];     // rows 0..62 of output_tensor
  float v = (tok != 0) ? emb[(size_t)tok * ESZ + e] : 0.0f;
  x[i] = (bf16_t)v;
}

// ---------------------------------------------------------------------------
// Zero-init recurrent state (d_ws is poisoned by the harness)
// ---------------------------------------------------------------------------
__global__ __launch_bounds__(256) void k_init_state(bf16_t* __restrict__ hbf,
                                                    float* __restrict__ c, int n) {
  int i = blockIdx.x * 256 + threadIdx.x;
  if (i < n) { hbf[i] = (bf16_t)0.0f; c[i] = 0.0f; }
}

// ---------------------------------------------------------------------------
// Fragment helpers (wave32 WMMA layouts, CDNA5 ISA §7.12.2):
//  A 16x32 bf16 : lane -> row M = lane%16 ; VGPR v holds K pair
//                 k(v) = (v<4 ? 2v : 16+2(v-4)) + 8*(lane/16)
//  B 32x16 bf16 : lane -> col N = lane%16 ; VGPR v holds K pair same as A
//  C/D 16x16 f32: lane -> col N = lane%16 ; VGPR r -> row M = r + 8*(lane/16)
// ---------------------------------------------------------------------------
union AFrag { v16bf v; u32 u[8]; };

__device__ inline v16bf load_a_lds(const bf16_t* __restrict__ rowbase, int half) {
  AFrag f;
#pragma unroll
  for (int v = 0; v < 8; ++v) {
    int k = (v < 4 ? 2 * v : 16 + 2 * (v - 4)) + 8 * half;
    f.u[v] = *(const u32*)(rowbase + k);
  }
  return f.v;
}

__device__ inline v16bf load_b_global(const bf16_t* __restrict__ wrow, int half) {
  AFrag f;
  const bf16_t* p = wrow;
#pragma unroll
  for (int v = 0; v < 8; ++v) f.u[v] = *(const u32*)(p + (v < 4 ? 2 * v : 16 + 2 * (v - 4)) + 8 * half);
  return f.v;
}

// ---------------------------------------------------------------------------
// LSTM gate GEMM:  gates(64 x 4096) = A0(64xK0) @ W0(4096xK0)^T
//                                   + A1(64xK1) @ W1(4096xK1)^T + bih + bhh
// Block = 256 threads = 8 waves; wave w owns N-tile n0 = bx*128 + 16w and all
// four M-tiles (batch 64). A chunks double-buffered through LDS via async
// copies; W streamed from L2 with prefetch. Grid = 4096/128 = 32 blocks.
// ---------------------------------------------------------------------------
__global__ __launch_bounds__(256) void k_lstm_gates(
    const bf16_t* __restrict__ A0, const bf16_t* __restrict__ W0, int K0,
    const bf16_t* __restrict__ A1, const bf16_t* __restrict__ W1, int K1,
    const float* __restrict__ bih, const float* __restrict__ bhh,
    float* __restrict__ gates) {
  __shared__ __align__(16) bf16_t sA[2][64 * 32];   // 2 x 4 KB K-chunk buffers

  const int tid  = threadIdx.x;
  const int wave = tid >> 5;
  const int lane = tid & 31;
  const int half = lane >> 4;
  const int l16  = lane & 15;
  const int n0   = blockIdx.x * 128 + wave * 16;
  const int srow = tid >> 2;      // staging: 64 rows ...
  const int sseg = tid & 3;       // ... x 4 b128 segments (one per thread)

  const int nc0 = K0 >> 5;
  const int nch = nc0 + (K1 >> 5);

  auto stage = [&](int c, int buf) {
    const bf16_t* A; int K, kc;
    if (c < nc0) { A = A0; K = K0; kc = c << 5; }
    else         { A = A1; K = K1; kc = (c - nc0) << 5; }
    lds_copy_b128(A + (size_t)srow * K + kc + sseg * 8,
                  &sA[buf][srow * 32 + sseg * 8]);
  };

  v8f acc[4] = {};                // 4 M-tiles of 16x16
  stage(0, 0);

  for (int c = 0; c < nch; ++c) {
    const int cur = c & 1;
    ASYNC_WAIT();                 // chunk c landed in LDS (this wave's issues)
    __syncthreads();              // ... visible to all waves; prior reads done
    if (c + 1 < nch) stage(c + 1, cur ^ 1);

    const bf16_t* W; int K, kc;
    if (c < nc0) { W = W0; K = K0; kc = c << 5; }
    else         { W = W1; K = K1; kc = (c - nc0) << 5; }
    const bf16_t* wrow = W + (size_t)(n0 + l16) * K + kc;
    __builtin_prefetch(wrow + 32, 0, 0);            // next K-chunk of W
    v16bf bfrag = load_b_global(wrow, half);

#pragma unroll
    for (int mt = 0; mt < 4; ++mt) {
      v16bf afrag = load_a_lds(&sA[cur][(mt * 16 + l16) * 32], half);
      acc[mt] = __builtin_amdgcn_wmma_f32_16x16x32_bf16(
          false, afrag, false, bfrag, (short)0, acc[mt], false, false);
    }
  }

  const float bsum = bih[n0 + l16] + bhh[n0 + l16];
#pragma unroll
  for (int mt = 0; mt < 4; ++mt)
#pragma unroll
    for (int r = 0; r < 8; ++r) {
      int m = mt * 16 + r + 8 * half;
      gates[(size_t)m * GSZ + n0 + l16] = acc[mt][r] + bsum;
    }
}

// ---------------------------------------------------------------------------
// LSTM pointwise: c' = sig(f)*c + sig(i)*tanh(g);  h' = sig(o)*tanh(c')
// ---------------------------------------------------------------------------
__global__ __launch_bounds__(256) void k_lstm_pointwise(
    const float* __restrict__ gates, float* __restrict__ c,
    bf16_t* __restrict__ hbf, bf16_t* __restrict__ hout) {
  int i = blockIdx.x * 256 + threadIdx.x;
  if (i >= BSZ * HSZ) return;
  int m = i >> 10, n = i & (HSZ - 1);
  const float* g = gates + (size_t)m * GSZ;
  float gi = g[n], gf = g[HSZ + n], gg = g[2 * HSZ + n], go = g[3 * HSZ + n];
  float si = 1.0f / (1.0f + __expf(-gi));
  float sf = 1.0f / (1.0f + __expf(-gf));
  float so = 1.0f / (1.0f + __expf(-go));
  float c2 = sf * c[i] + si * tanhf(gg);
  float h2 = so * tanhf(c2);
  c[i] = c2;
  hbf[i] = (bf16_t)h2;
  if (hout) hout[i] = (bf16_t)h2;
}

// ---------------------------------------------------------------------------
// Fused projection + split-V log-softmax stats + argmax + target-logit.
// Never materializes the 516 MB logits tensor. Block = 16 rows x 512 cols;
// 8 waves, wave w owns cols [64w, 64w+64) as 4 WMMA N-tiles; K-loop over 1024
// with double-buffered async A staging. Tile logits spill to LDS (32 KB),
// then per-row reduce -> partial {max, sum-exp@localmax, argmax}.
// ---------------------------------------------------------------------------
__global__ __launch_bounds__(256) void k_proj(
    const bf16_t* __restrict__ Hb,    // ROWS x 1024
    const bf16_t* __restrict__ Wb,    // VSZ x 1024 (bf16)
    const float* __restrict__ bout,   // VSZ
    const int* __restrict__ tokens,   // (T,B) flat int32
    float* __restrict__ pmax, float* __restrict__ psum,
    int* __restrict__ pidx, float* __restrict__ tl) {
  __shared__ __align__(16) bf16_t sA[2][16 * 32];   // 2 x 1 KB chunk buffers
  __shared__ float sL[16 * PROJ_COLS];              // 32 KB logits tile

  const int tid  = threadIdx.x;
  const int wave = tid >> 5;
  const int lane = tid & 31;
  const int half = lane >> 4;
  const int l16  = lane & 15;
  const int r0   = blockIdx.x * 16;
  const int c0   = blockIdx.y * PROJ_COLS;
  const int n0   = c0 + wave * 64;
  const int srow = tid >> 2;      // staging threads 0..63: 16 rows x 4 segs
  const int sseg = tid & 3;

  auto stage = [&](int kc, int buf) {
    if (tid < 64)
      lds_copy_b128(Hb + (size_t)(r0 + srow) * HSZ + kc + sseg * 8,
                    &sA[buf][srow * 32 + sseg * 8]);
  };

  v8f acc[4] = {};                           // 4 N-tiles of 16x16
  stage(0, 0);

  for (int kc = 0; kc < HSZ; kc += 32) {
    const int cur = (kc >> 5) & 1;
    ASYNC_WAIT();
    __syncthreads();
    if (kc + 32 < HSZ) stage(kc + 32, cur ^ 1);

    v16bf afrag = load_a_lds(&sA[cur][l16 * 32], half);
#pragma unroll
    for (int nt = 0; nt < 4; ++nt) {
      int n   = n0 + nt * 16 + l16;
      int nld = (n < VSZ) ? n : (VSZ - 1);   // clamp OOB tail rows
      const bf16_t* wrow = Wb + (size_t)nld * HSZ + kc;
      __builtin_prefetch(wrow + 32, 0, 0);
      v16bf bfrag = load_b_global(wrow, half);
      acc[nt] = __builtin_amdgcn_wmma_f32_16x16x32_bf16(
          false, afrag, false, bfrag, (short)0, acc[nt], false, false);
    }
  }

  // spill tile logits (+bias, -inf past V) to LDS
#pragma unroll
  for (int nt = 0; nt < 4; ++nt) {
    int col = wave * 64 + nt * 16 + l16;
    int n   = c0 + col;
    float bias = (n < VSZ) ? bout[n] : 0.0f;
#pragma unroll
    for (int r = 0; r < 8; ++r) {
      int m = r + 8 * half;
      float val = (n < VSZ) ? (acc[nt][r] + bias) : -__builtin_inff();
      sL[m * PROJ_COLS + col] = val;
    }
  }
  __syncthreads();

  // per-row reduce over 512 cols: 16 lanes per row (tid>>4 = row in tile)
  const int row = tid >> 4;
  const int lr  = tid & 15;
  float mx = -__builtin_inff();
  int   mi = 0x7fffffff;
  for (int ccc = lr; ccc < PROJ_COLS; ccc += 16) {
    float v = sL[row * PROJ_COLS + ccc];
    if (v > mx) { mx = v; mi = c0 + ccc; }
  }
#pragma unroll
  for (int off = 8; off; off >>= 1) {
    float om = __shfl_xor(mx, off, 16);
    int   oi = __shfl_xor(mi, off, 16);
    if (om > mx || (om == mx && oi < mi)) { mx = om; mi = oi; }
  }
  float s = 0.0f;
  for (int ccc = lr; ccc < PROJ_COLS; ccc += 16)
    s += __expf(sL[row * PROJ_COLS + ccc] - mx);
#pragma unroll
  for (int off = 8; off; off >>= 1) s += __shfl_xor(s, off, 16);

  if (lr == 0) {
    int r = r0 + row;
    pmax[(size_t)r * NCB + blockIdx.y] = mx;
    psum[(size_t)r * NCB + blockIdx.y] = s;
    pidx[(size_t)r * NCB + blockIdx.y] = mi;
    int t = r >> 6, b = r & 63;
    int tgt = tokens[(t + 1) * BSZ + b];
    if (tgt >= c0 && tgt < c0 + PROJ_COLS)
      tl[r] = sL[row * PROJ_COLS + (tgt - c0)];
  }
}

// ---------------------------------------------------------------------------
// Combine split-V partials per row: global logsumexp, NLL, argmax; write preds.
// ---------------------------------------------------------------------------
__global__ __launch_bounds__(256) void k_finalize(
    const float* __restrict__ pmax, const float* __restrict__ psum,
    const int* __restrict__ pidx, const float* __restrict__ tl,
    const int* __restrict__ tokens,
    float* __restrict__ nllv, float* __restrict__ validv,
    float* __restrict__ dout) {
  int r = blockIdx.x * 256 + threadIdx.x;
  if (r < BSZ) dout[1 + r] = 2.0f;           // result[0, :] = 2
  if (r >= ROWS) return;

  float M = -__builtin_inff();
  float bestv = -__builtin_inff();
  int   best = 0;
  for (int cb = 0; cb < NCB; ++cb) {
    float m = pmax[(size_t)r * NCB + cb];
    if (m > M) M = m;
    if (m > bestv) { bestv = m; best = pidx[(size_t)r * NCB + cb]; }
  }
  float S = 0.0f;
  for (int cb = 0; cb < NCB; ++cb)
    S += psum[(size_t)r * NCB + cb] * __expf(pmax[(size_t)r * NCB + cb] - M);
  float lse = M + __logf(S);
  float nll = lse - tl[r];

  int t = r >> 6, b = r & 63;
  int tgt = tokens[(t + 1) * BSZ + b];
  float valid = (tgt != 1) ? 1.0f : 0.0f;
  nllv[r]   = nll * valid;
  validv[r] = valid;
  dout[1 + (t + 1) * BSZ + b] = (float)best;   // result[1+t, b]
}

// ---------------------------------------------------------------------------
// loss = sum_t [ sum_b(nll*valid) / max(sum_b valid, 1) ] / T
// ---------------------------------------------------------------------------
__global__ __launch_bounds__(64) void k_loss(const float* __restrict__ nllv,
                                             const float* __restrict__ validv,
                                             float* __restrict__ dout) {
  __shared__ float sce[64];
  int t = threadIdx.x;
  float ce = 0.0f;
  if (t < TSTEPS) {
    float s = 0.0f, c = 0.0f;
    for (int b = 0; b < BSZ; ++b) { s += nllv[t * BSZ + b]; c += validv[t * BSZ + b]; }
    ce = s / fmaxf(c, 1.0f);
  }
  sce[t] = ce;
  __syncthreads();
  if (t == 0) {
    float tot = 0.0f;
    for (int i = 0; i < 64; ++i) tot += sce[i];
    dout[0] = tot / 64.0f;    // / T
  }
}

// ---------------------------------------------------------------------------
// Host orchestration. Workspace (~107 MB): bf16 weights + activations +
// split-V partials. Everything launched on `stream`; no mallocs/syncs.
// ---------------------------------------------------------------------------
extern "C" void kernel_launch(void* const* d_in, const int* in_sizes, int n_in,
                              void* d_out, int out_size, void* d_ws, size_t ws_size,
                              hipStream_t stream) {
  (void)in_sizes; (void)n_in; (void)out_size; (void)ws_size;

  const int*   tokens = (const int*)  d_in[0];   // (T,B) int32
  const float* emb    = (const float*)d_in[7];   // (V,E)
  const float* W_ih0  = (const float*)d_in[8];   // (4H,E)
  const float* W_hh0  = (const float*)d_in[9];   // (4H,H)
  const float* b_ih0  = (const float*)d_in[10];
  const float* b_hh0  = (const float*)d_in[11];
  const float* W_ih1  = (const float*)d_in[12];  // (4H,H)
  const float* W_hh1  = (const float*)d_in[13];  // (4H,H)
  const float* b_ih1  = (const float*)d_in[14];
  const float* b_hh1  = (const float*)d_in[15];
  const float* W_out  = (const float*)d_in[16];  // (V,H)
  const float* b_out  = (const float*)d_in[17];
  float* dout = (float*)d_out;                   // [loss | result(64x64)]

  char* ws = (char*)d_ws;
  size_t off = 0;
  auto take = [&](size_t bytes) -> char* {
    char* p = ws + off;
    off = (off + bytes + 255) & ~(size_t)255;
    return p;
  };

  bf16_t* wih0_bf = (bf16_t*)take((size_t)GSZ * ESZ * 2);
  bf16_t* whh0_bf = (bf16_t*)take((size_t)GSZ * HSZ * 2);
  bf16_t* wih1_bf = (bf16_t*)take((size_t)GSZ * HSZ * 2);
  bf16_t* whh1_bf = (bf16_t*)take((size_t)GSZ * HSZ * 2);
  bf16_t* wout_bf = (bf16_t*)take((size_t)VSZ * HSZ * 2);
  bf16_t* x_bf    = (bf16_t*)take((size_t)ROWS * ESZ * 2);
  bf16_t* hout_bf = (bf16_t*)take((size_t)ROWS * HSZ * 2);
  bf16_t* h0_bf   = (bf16_t*)take((size_t)BSZ * HSZ * 2);
  bf16_t* h1_bf   = (bf16_t*)take((size_t)BSZ * HSZ * 2);
  float*  c01     = (float*) take((size_t)2 * BSZ * HSZ * 4);
  float*  gates   = (float*) take((size_t)BSZ * GSZ * 4);
  float*  pmax    = (float*) take((size_t)ROWS * NCB * 4);
  float*  psum    = (float*) take((size_t)ROWS * NCB * 4);
  int*    pidx    = (int*)   take((size_t)ROWS * NCB * 4);
  float*  tl      = (float*) take((size_t)ROWS * 4);
  float*  nllv    = (float*) take((size_t)ROWS * 4);
  float*  validv  = (float*) take((size_t)ROWS * 4);
  float*  c0 = c01;
  float*  c1 = c01 + (size_t)BSZ * HSZ;

  auto cast = [&](const float* s, bf16_t* d, size_t n) {
    k_cast_bf16<<<(unsigned)((n + 255) / 256), 256, 0, stream>>>(s, d, (int)n);
  };
  cast(W_ih0, wih0_bf, (size_t)GSZ * ESZ);
  cast(W_hh0, whh0_bf, (size_t)GSZ * HSZ);
  cast(W_ih1, wih1_bf, (size_t)GSZ * HSZ);
  cast(W_hh1, whh1_bf, (size_t)GSZ * HSZ);
  cast(W_out, wout_bf, (size_t)VSZ * HSZ);

  {
    size_t n = (size_t)ROWS * ESZ;
    k_embed<<<(unsigned)((n + 255) / 256), 256, 0, stream>>>(tokens, emb, x_bf, (int)n);
  }
  k_init_state<<<(2 * BSZ * HSZ + 255) / 256, 256, 0, stream>>>(h0_bf, c0, 2 * BSZ * HSZ);

  // ---- recurrence: 63 sequential steps, 2 layers ----
  for (int t = 0; t < TSTEPS; ++t) {
    const bf16_t* xt = x_bf + (size_t)t * BSZ * ESZ;
    // layer 0: gates = x_t@Wih0^T + h0@Whh0^T + b
    k_lstm_gates<<<GSZ / 128, 256, 0, stream>>>(xt, wih0_bf, ESZ,
                                                h0_bf, whh0_bf, HSZ,
                                                b_ih0, b_hh0, gates);
    k_lstm_pointwise<<<(BSZ * HSZ) / 256, 256, 0, stream>>>(gates, c0, h0_bf, nullptr);
    // layer 1: gates = h0@Wih1^T + h1@Whh1^T + b
    k_lstm_gates<<<GSZ / 128, 256, 0, stream>>>(h0_bf, wih1_bf, HSZ,
                                                h1_bf, whh1_bf, HSZ,
                                                b_ih1, b_hh1, gates);
    k_lstm_pointwise<<<(BSZ * HSZ) / 256, 256, 0, stream>>>(
        gates, c1, h1_bf, hout_bf + (size_t)t * BSZ * HSZ);
  }

  // ---- fused projection + log-softmax stats (split over V) ----
  k_proj<<<dim3(ROWS / 16, NCB), 256, 0, stream>>>(hout_bf, wout_bf, b_out, tokens,
                                                   pmax, psum, pidx, tl);
  k_finalize<<<(ROWS + 255) / 256 + 1, 256, 0, stream>>>(pmax, psum, pidx, tl, tokens,
                                                         nllv, validv, dout);
  k_loss<<<1, 64, 0, stream>>>(nllv, validv, dout);
}
